// PointNet2Encoder_27487790694914
// MI455X (gfx1250) — compile-verified, matrix operand order checked by reading, not run on hardware
//
#include <hip/hip_runtime.h>
#include <hip/hip_bf16.h>

typedef __attribute__((ext_vector_type(8)))  _Float16 v8h;
typedef __attribute__((ext_vector_type(16))) _Float16 v16h;
typedef __attribute__((ext_vector_type(8)))  float    v8f;

#define BN_EPS 1e-5f

// ---------------------------------------------------------------------------
// 0. transpose xyz (B,3,N) -> pts (B,N,3)
// ---------------------------------------------------------------------------
__global__ void pn2_transpose_kernel(const float* __restrict__ xyz,
                                     float* __restrict__ pts,
                                     int B, int N) {
    int i = blockIdx.x * blockDim.x + threadIdx.x;
    int total = B * N * 3;
    if (i >= total) return;
    int b = i / (N * 3);
    int r = i - b * N * 3;
    int n = r / 3;
    int c = r - n * 3;
    pts[i] = xyz[(size_t)b * 3 * N + (size_t)c * N + n];
}

// ---------------------------------------------------------------------------
// 1. Farthest point sampling: one block per batch, serial over S samples.
//    pts: (B,N,3). Writes idx (B,S) and sampled coords (B,S,3).
// ---------------------------------------------------------------------------
__global__ __launch_bounds__(256) void pn2_fps_kernel(
    const float* __restrict__ pts, int N, int S,
    int* __restrict__ out_idx, float* __restrict__ out_xyz) {
    int b = blockIdx.x;
    int t = threadIdx.x;
    const float* P = pts + (size_t)b * N * 3;

    __shared__ float s_dist[4096];
    __shared__ float s_bv[256];
    __shared__ int   s_bi[256];
    __shared__ int   s_far;

    for (int i = t; i < N; i += 256) s_dist[i] = 1e10f;
    if (t == 0) s_far = 0;
    __syncthreads();

    for (int s = 0; s < S; ++s) {
        int far = s_far;
        if (t == 0) {
            out_idx[b * S + s] = far;
            out_xyz[((size_t)b * S + s) * 3 + 0] = P[far * 3 + 0];
            out_xyz[((size_t)b * S + s) * 3 + 1] = P[far * 3 + 1];
            out_xyz[((size_t)b * S + s) * 3 + 2] = P[far * 3 + 2];
        }
        float cx = P[far * 3 + 0], cy = P[far * 3 + 1], cz = P[far * 3 + 2];
        float bv = -1.0f; int bi = 0x7fffffff;
        for (int i = t; i < N; i += 256) {
            float dx = P[i * 3 + 0] - cx;
            float dy = P[i * 3 + 1] - cy;
            float dz = P[i * 3 + 2] - cz;
            float d = dx * dx + dy * dy + dz * dz;
            float nd = fminf(s_dist[i], d);
            s_dist[i] = nd;
            if (nd > bv || (nd == bv && i < bi)) { bv = nd; bi = i; }
        }
        s_bv[t] = bv; s_bi[t] = bi;
        __syncthreads();
        for (int off = 128; off > 0; off >>= 1) {
            if (t < off) {
                float ov = s_bv[t + off]; int oi = s_bi[t + off];
                if (ov > s_bv[t] || (ov == s_bv[t] && oi < s_bi[t])) {
                    s_bv[t] = ov; s_bi[t] = oi;
                }
            }
            __syncthreads();
        }
        if (t == 0) s_far = s_bi[0];
        __syncthreads();
    }
}

// ---------------------------------------------------------------------------
// 2. Fold BN into weights: Wf[o][k] (f16, K padded) , bf[o] (f32)
// ---------------------------------------------------------------------------
__global__ void pn2_fold_kernel(const float* __restrict__ W,
                                const float* __restrict__ bc,
                                const float* __restrict__ g,
                                const float* __restrict__ bt,
                                const float* __restrict__ m,
                                const float* __restrict__ vv,
                                _Float16* __restrict__ Wf,
                                float* __restrict__ bf,
                                int cout, int cin, int Kpad) {
    int i = blockIdx.x * blockDim.x + threadIdx.x;
    int total = cout * Kpad;
    if (i >= total) return;
    int o = i / Kpad;
    int c = i - o * Kpad;
    float scale = g[o] * rsqrtf(vv[o] + BN_EPS);
    Wf[i] = (_Float16)((c < cin) ? W[(size_t)o * cin + c] * scale : 0.0f);
    if (c == 0) bf[o] = (bc[o] - m[o]) * scale + bt[o];
}

// ---------------------------------------------------------------------------
// 3. Ball-query + group: one thread per centroid. Scan points in index order,
//    keep first NS within r^2 (== reference sort-then-truncate), pad w/ first.
//    Row layout: X[((b*S+s)*NS + k) * Kpad]: [0..2]=delta xyz, [3..3+C)=feat, 0-pad.
// ---------------------------------------------------------------------------
__device__ __forceinline__ void pn2_write_row(
    _Float16* dst, const float* P, const float* feats,
    int b, int n, int N, int featC, int Kpad,
    float cx, float cy, float cz) {
    dst[0] = (_Float16)(P[n * 3 + 0] - cx);
    dst[1] = (_Float16)(P[n * 3 + 1] - cy);
    dst[2] = (_Float16)(P[n * 3 + 2] - cz);
    if (feats) {
        const float* f = feats + ((size_t)b * N + n) * featC;
        for (int c = 0; c < featC; ++c) dst[3 + c] = (_Float16)f[c];
    }
    for (int c = 3 + featC; c < Kpad; ++c) dst[c] = (_Float16)0.0f;
}

__global__ void pn2_group_kernel(const float* __restrict__ pts,
                                 const float* __restrict__ feats,
                                 const float* __restrict__ new_xyz,
                                 _Float16* __restrict__ X,
                                 int total, int N, int S, int NS,
                                 int featC, int Kpad, float r2) {
    int i = blockIdx.x * blockDim.x + threadIdx.x;
    if (i >= total) return;
    int b = i / S;
    const float* P = pts + (size_t)b * N * 3;
    float cx = new_xyz[(size_t)i * 3 + 0];
    float cy = new_xyz[(size_t)i * 3 + 1];
    float cz = new_xyz[(size_t)i * 3 + 2];
    int count = 0, first = 0;
    for (int n = 0; n < N; ++n) {
        float dx = P[n * 3 + 0] - cx;
        float dy = P[n * 3 + 1] - cy;
        float dz = P[n * 3 + 2] - cz;
        float d = dx * dx + dy * dy + dz * dz;
        if (d <= r2) {
            if (count == 0) first = n;
            _Float16* dst = X + ((size_t)i * NS + count) * Kpad;
            pn2_write_row(dst, P, feats, b, n, N, featC, Kpad, cx, cy, cz);
            if (++count == NS) break;
        }
    }
    for (int k = count; k < NS; ++k) {
        _Float16* dst = X + ((size_t)i * NS + k) * Kpad;
        pn2_write_row(dst, P, feats, b, first, N, featC, Kpad, cx, cy, cz);
    }
}

// ---------------------------------------------------------------------------
// 4. WMMA GEMM: Y[M,N] = relu(X[M,K] @ Wf[N,K]^T + bf), f16 in/out, f32 acc.
//    One wave computes a 16x16 tile. K multiple of 32, M,N multiples of 16.
// ---------------------------------------------------------------------------
__global__ __launch_bounds__(256) void pn2_gemm_bias_relu_kernel(
    const _Float16* __restrict__ X, const _Float16* __restrict__ W,
    const float* __restrict__ bias, _Float16* __restrict__ Y,
    int M, int N, int K) {
    int gthread = blockIdx.x * blockDim.x + threadIdx.x;
    int wave = gthread >> 5;
    int lane = threadIdx.x & 31;
    int tilesN = N >> 4;
    int tilesM = M >> 4;
    if (wave >= tilesM * tilesN) return;        // wave-uniform: EXEC stays all-1
    int tm = wave / tilesN;
    int tn = wave - tm * tilesN;
    int half = lane >> 4;                        // 0 or 1
    int l15  = lane & 15;

    // 16-bit A layout: lanes 0-15 hold row M=lane, K {0..7,16..23};
    // lanes 16-31 hold row M=lane-16, K {8..15,24..31}. B mirrored (B^T rows).
    const _Float16* xrow = X + (size_t)(tm * 16 + l15) * K + half * 8;
    const _Float16* wrow = W + (size_t)(tn * 16 + l15) * K + half * 8;

    v8f acc = {};
    for (int k0 = 0; k0 < K; k0 += 32) {
        v8h alo = *(const v8h*)(xrow + k0);
        v8h ahi = *(const v8h*)(xrow + k0 + 16);
        v8h blo = *(const v8h*)(wrow + k0);
        v8h bhi = *(const v8h*)(wrow + k0 + 16);
        v16h a, bm;
#pragma unroll
        for (int j = 0; j < 8; ++j) {
            a[j] = alo[j]; a[j + 8] = ahi[j];
            bm[j] = blo[j]; bm[j + 8] = bhi[j];
        }
        acc = __builtin_amdgcn_wmma_f32_16x16x32_f16(
            false, a, false, bm, (short)0, acc, false, false);
    }

    // C/D layout: lanes 0-15: col=lane, rows 0..7; lanes 16-31: col=lane-16, rows 8..15
    int col = tn * 16 + l15;
    float bs = bias[col];
    int mbase = tm * 16 + half * 8;
#pragma unroll
    for (int r = 0; r < 8; ++r) {
        float v = acc[r] + bs;
        v = v > 0.0f ? v : 0.0f;
        Y[(size_t)(mbase + r) * N + col] = (_Float16)v;
    }
}

// ---------------------------------------------------------------------------
// 5. Max-pool over group dim: Y (groups*G, N) f16 -> P (groups, N) f32
// ---------------------------------------------------------------------------
__global__ void pn2_maxpool_kernel(const _Float16* __restrict__ Y,
                                   float* __restrict__ P,
                                   int groups, int N, int G) {
    int i = blockIdx.x * blockDim.x + threadIdx.x;
    if (i >= groups * N) return;
    int g = i / N;
    int c = i - g * N;
    const _Float16* p = Y + (size_t)g * G * N + c;
    float m = (float)p[0];
    for (int k = 1; k < G; ++k) m = fmaxf(m, (float)p[(size_t)k * N]);
    P[i] = m;
}

// ---------------------------------------------------------------------------
// 6. Build sa3 input: X3[row, 0..2]=l2_xyz, [3..258]=l2_pts, rest 0 (f16)
// ---------------------------------------------------------------------------
__global__ void pn2_build3_kernel(const float* __restrict__ l2_xyz,
                                  const float* __restrict__ l2_pts,
                                  _Float16* __restrict__ X,
                                  int rows, int C, int Kpad) {
    int i = blockIdx.x * blockDim.x + threadIdx.x;
    if (i >= rows * Kpad) return;
    int r = i / Kpad;
    int c = i - r * Kpad;
    float v = 0.0f;
    if (c < 3)            v = l2_xyz[(size_t)r * 3 + c];
    else if (c < 3 + C)   v = l2_pts[(size_t)r * C + (c - 3)];
    X[i] = (_Float16)v;
}

// ---------------------------------------------------------------------------
// 7. L2 normalize rows of (B,1024) -> d_out (f32)
// ---------------------------------------------------------------------------
__global__ __launch_bounds__(256) void pn2_norm_kernel(
    const float* __restrict__ P, float* __restrict__ out, int C) {
    int b = blockIdx.x;
    int t = threadIdx.x;
    __shared__ float red[256];
    float s = 0.0f;
    for (int c = t; c < C; c += 256) {
        float v = P[(size_t)b * C + c];
        s += v * v;
    }
    red[t] = s;
    __syncthreads();
    for (int off = 128; off > 0; off >>= 1) {
        if (t < off) red[t] += red[t + off];
        __syncthreads();
    }
    float inv = 1.0f / fmaxf(sqrtf(red[0]), 1e-12f);
    for (int c = t; c < C; c += 256) out[(size_t)b * C + c] = P[(size_t)b * C + c] * inv;
}

// ===========================================================================
extern "C" void kernel_launch(void* const* d_in, const int* in_sizes, int n_in,
                              void* d_out, int out_size, void* d_ws, size_t ws_size,
                              hipStream_t stream) {
    (void)in_sizes; (void)n_in; (void)out_size; (void)ws_size;
    const int B = 16, N1 = 4096, S1 = 512, NS1 = 32;
    const int S2 = 128, NS2 = 64;
    const float R1SQ = 0.2f * 0.2f, R2SQ = 0.4f * 0.4f;
    const int K1 = 32, C1A = 64, C1B = 64, C1C = 128;
    const int K2 = 160, C2A = 128, C2B = 128, C2C = 256;
    const int K3 = 288, C3A = 256, C3B = 512, C3C = 1024;
    const int M1 = B * S1 * NS1;   // 262144
    const int M2 = B * S2 * NS2;   // 131072
    const int M3 = B * S2;         // 2048

    const float* xyz = (const float*)d_in[0];

    // ------- workspace bump allocator -------
    char* ws = (char*)d_ws;
    size_t off = 0;
    auto alloc = [&](size_t bytes) -> void* {
        void* p = ws + off;
        off = (off + bytes + 255) & ~(size_t)255;
        return p;
    };
    float*    pts1  = (float*)alloc((size_t)B * N1 * 3 * 4);
    int*      fidx1 = (int*)  alloc((size_t)B * S1 * 4);
    float*    nxyz1 = (float*)alloc((size_t)B * S1 * 3 * 4);
    float*    pool1 = (float*)alloc((size_t)B * S1 * C1C * 4);
    int*      fidx2 = (int*)  alloc((size_t)B * S2 * 4);
    float*    nxyz2 = (float*)alloc((size_t)B * S2 * 3 * 4);
    float*    pool2 = (float*)alloc((size_t)B * S2 * C2C * 4);
    float*    pool3 = (float*)alloc((size_t)B * C3C * 4);

    _Float16* Wf[9]; float* bf[9];
    const int couts[9] = {C1A, C1B, C1C, C2A, C2B, C2C, C3A, C3B, C3C};
    const int cins [9] = {3, 64, 64, 131, 128, 128, 259, 256, 512};
    const int kpads[9] = {K1, 64, 64, K2, 128, 128, K3, 256, 512};
    for (int l = 0; l < 9; ++l) {
        Wf[l] = (_Float16*)alloc((size_t)couts[l] * kpads[l] * 2);
        bf[l] = (float*)   alloc((size_t)couts[l] * 4);
    }
    // ping-pong activation buffers
    _Float16* BUFA = (_Float16*)alloc((size_t)M2 * K2 * 2);   // 40 MB max use
    _Float16* BUFB = (_Float16*)alloc((size_t)M1 * C1C * 2);  // 64 MB max use

    // ------- fold weights -------
    for (int l = 0; l < 9; ++l) {
        const float* W  = (const float*)d_in[1 + l * 6 + 0];
        const float* bb = (const float*)d_in[1 + l * 6 + 1];
        const float* g  = (const float*)d_in[1 + l * 6 + 2];
        const float* bt = (const float*)d_in[1 + l * 6 + 3];
        const float* m  = (const float*)d_in[1 + l * 6 + 4];
        const float* v  = (const float*)d_in[1 + l * 6 + 5];
        int total = couts[l] * kpads[l];
        pn2_fold_kernel<<<(total + 255) / 256, 256, 0, stream>>>(
            W, bb, g, bt, m, v, Wf[l], bf[l], couts[l], cins[l], kpads[l]);
    }

    auto gemm = [&](const _Float16* X, int l, _Float16* Y, int M) {
        int tiles = (M >> 4) * (couts[l] >> 4);
        int threads = tiles * 32;
        pn2_gemm_bias_relu_kernel<<<(threads + 255) / 256, 256, 0, stream>>>(
            X, Wf[l], bf[l], Y, M, couts[l], kpads[l]);
    };

    // ------- level 1 -------
    {
        int total = B * N1 * 3;
        pn2_transpose_kernel<<<(total + 255) / 256, 256, 0, stream>>>(xyz, pts1, B, N1);
    }
    pn2_fps_kernel<<<B, 256, 0, stream>>>(pts1, N1, S1, fidx1, nxyz1);
    {
        int total = B * S1;
        pn2_group_kernel<<<(total + 255) / 256, 256, 0, stream>>>(
            pts1, nullptr, nxyz1, BUFA, total, N1, S1, NS1, 0, K1, R1SQ);
    }
    gemm(BUFA, 0, BUFB, M1);
    gemm(BUFB, 1, BUFA, M1);
    gemm(BUFA, 2, BUFB, M1);
    {
        int total = B * S1 * C1C;
        pn2_maxpool_kernel<<<(total + 255) / 256, 256, 0, stream>>>(
            BUFB, pool1, B * S1, C1C, NS1);
    }

    // ------- level 2 -------
    pn2_fps_kernel<<<B, 256, 0, stream>>>(nxyz1, S1, S2, fidx2, nxyz2);
    {
        int total = B * S2;
        pn2_group_kernel<<<(total + 255) / 256, 256, 0, stream>>>(
            nxyz1, pool1, nxyz2, BUFA, total, S1, S2, NS2, C1C, K2, R2SQ);
    }
    gemm(BUFA, 3, BUFB, M2);
    gemm(BUFB, 4, BUFA, M2);
    gemm(BUFA, 5, BUFB, M2);
    {
        int total = B * S2 * C2C;
        pn2_maxpool_kernel<<<(total + 255) / 256, 256, 0, stream>>>(
            BUFB, pool2, B * S2, C2C, NS2);
    }

    // ------- level 3 (group-all) -------
    {
        int total = M3 * K3;
        pn2_build3_kernel<<<(total + 255) / 256, 256, 0, stream>>>(
            nxyz2, pool2, BUFA, M3, C2C, K3);
    }
    gemm(BUFA, 6, BUFB, M3);
    gemm(BUFB, 7, BUFA, M3);
    gemm(BUFA, 8, BUFB, M3);
    {
        int total = B * C3C;
        pn2_maxpool_kernel<<<(total + 255) / 256, 256, 0, stream>>>(
            BUFB, pool3, B, C3C, S2);
    }

    // ------- normalize -------
    pn2_norm_kernel<<<B, 256, 0, stream>>>(pool3, (float*)d_out, C3C);
}
